// GAT_88871463289130
// MI455X (gfx1250) — matrix operand order repaired and checked by reading, not compiled, not run on hardware
//
#include <hip/hip_runtime.h>
#include <hip/hip_bf16.h>

#define NNODE 50000
#define NEDGE 800000
#define FIN   128
#define NH    8
#define NC_   16          // channels per head
#define DDIM  128         // NH * NC_
#define NG    64
#define NCLS  10
#define NLAY  2
#define NEG_SLOPE 0.2f

#define ROW_TILES (NNODE / 16)   // 3125 (exact)
#define WPB 8                    // waves per block in GEMM

typedef __attribute__((ext_vector_type(16))) __bf16 v16bf;
typedef __attribute__((ext_vector_type(8)))  float  v8f;

// ---------- float <-> order-preserving uint (for atomic max on floats) ----------
__host__ __device__ inline unsigned f2ord(float f) {
    unsigned u = __builtin_bit_cast(unsigned, f);
    return (u & 0x80000000u) ? ~u : (u | 0x80000000u);
}
__host__ __device__ inline float ord2f(unsigned o) {
    unsigned u = (o & 0x80000000u) ? (o ^ 0x80000000u) : ~o;
    return __builtin_bit_cast(float, u);
}

// ---------------------------------------------------------------------------
// GEMM: Out[M x 128] = A[M x 128] @ W[128 x 128] (+ bias).
// 256-thread block = 8 waves. Block stages the whole W as packed bf16 in LDS
// (32 KB), laid out so each lane's B-fragment is 32 contiguous bytes:
//   flat bf16 index  ((kk*2 + half)*128 + col)*16 + e   <=>  K = (kk*2+half)*16+e
// Each wave then computes one 16-row x 128-col strip: per kk it loads the
// A-fragment once from global f32 (converted to bf16) and reuses it against
// the 8 B-fragments read via ds_load_b128 -> 32 v_wmma per wave.
// Fragment layouts used by this kernel (contiguous K order, wave32):
//   A (16x32 bf16): lane&15 = M, half = lane>>4; e -> K = half*16 + e
//   B (32x16 bf16): lane&15 = N; e -> K = half*16 + e
//   C/D (16x16 f32): acc[r] = (M = half*8 + r, N = lane&15)
// ---------------------------------------------------------------------------
__global__ __launch_bounds__(256)
void gemm128_wmma_kernel(const float* __restrict__ A,
                         const float* __restrict__ W,
                         const float* __restrict__ bias,
                         float* __restrict__ Out)
{
    __shared__ __align__(32) __bf16 lds_w[4 * 2 * 128 * 16];   // 16384 bf16 = 32 KB

    // ---- cooperative fill: pack + convert W into B-fragment layout ----
    const int tid = threadIdx.x;
    for (int i = tid; i < 8192; i += 256) {          // i indexes bf16 pairs
        const int e   = (2 * i) & 15;                // even, pair shares col/kh
        const int col = ((2 * i) >> 4) & 127;
        const int kh  = (2 * i) >> 11;               // kk*2 + half, 0..7
        const int k   = kh * 16 + e;
        lds_w[2 * i]     = (__bf16)W[(size_t)k       * DDIM + col];
        lds_w[2 * i + 1] = (__bf16)W[(size_t)(k + 1) * DDIM + col];
    }
    __syncthreads();

    const int wave   = tid >> 5;
    const int lane   = tid & 31;
    const int tile_m = blockIdx.x * WPB + wave;      // wave-uniform
    if (tile_m >= ROW_TILES) return;                 // whole-wave exit only

    const int m    = lane & 15;
    const int half = lane >> 4;

    v8f acc[8];
#pragma unroll
    for (int tn = 0; tn < 8; ++tn) {
        const float bv = bias ? bias[tn * 16 + m] : 0.0f;
#pragma unroll
        for (int r = 0; r < 8; ++r) acc[tn][r] = bv;
    }

    const float* arow = A + (size_t)(tile_m * 16 + m) * DDIM;

#pragma unroll
    for (int kk = 0; kk < 4; ++kk) {
        const int kbase = kk * 32;
        v16bf afrag;
#pragma unroll
        for (int e = 0; e < 16; ++e) {
            const int ka = kbase + half * 16 + e;
            afrag[e] = (__bf16)arow[ka];
        }
#pragma unroll
        for (int tn = 0; tn < 8; ++tn) {
            const v16bf bfrag =
                *(const v16bf*)&lds_w[(((kk * 2 + half) * 128) + tn * 16 + m) * 16];
            acc[tn] = __builtin_amdgcn_wmma_f32_16x16x32_bf16(
                          false, afrag, false, bfrag, (short)0, acc[tn], false, false);
        }
    }

#pragma unroll
    for (int tn = 0; tn < 8; ++tn) {
#pragma unroll
        for (int r = 0; r < 8; ++r) {
            const int row = tile_m * 16 + half * 8 + r;
            Out[(size_t)row * DDIM + tn * 16 + m] = acc[tn][r];
        }
    }
}

// a_src[n,h] = <hp[n,h,:], att_src[h,:]>, same for a_dst
__global__ void att_kernel(const float* __restrict__ hp,
                           const float* __restrict__ asrc_w,
                           const float* __restrict__ adst_w,
                           float* __restrict__ a_src,
                           float* __restrict__ a_dst)
{
    int t = blockIdx.x * blockDim.x + threadIdx.x;
    if (t >= NNODE * NH) return;
    const int node = t / NH, head = t % NH;
    const float* v  = hp + (size_t)node * DDIM + head * NC_;
    const float* ws = asrc_w + head * NC_;
    const float* wd = adst_w + head * NC_;
    float s = 0.f, d = 0.f;
#pragma unroll
    for (int c = 0; c < NC_; ++c) { s += v[c] * ws[c]; d += v[c] * wd[c]; }
    a_src[t] = s; a_dst[t] = d;
}

__global__ void fill_u32_kernel(unsigned* __restrict__ p, unsigned val, int n)
{
    int t = blockIdx.x * blockDim.x + threadIdx.x;
    if (t < n) p[t] = val;
}

__device__ __forceinline__ float edge_logit(const float* a_src, const float* a_dst,
                                            int s, int d, int h)
{
    float v = a_src[s * NH + h] + a_dst[d * NH + h];
    return v > 0.f ? v : NEG_SLOPE * v;      // LeakyReLU
}

__global__ void edge_max_kernel(const int* __restrict__ ei,
                                const float* __restrict__ a_src,
                                const float* __restrict__ a_dst,
                                unsigned* __restrict__ nmax)
{
    int t = blockIdx.x * blockDim.x + threadIdx.x;
    if (t >= NEDGE * NH) return;
    const int e = t / NH, h = t % NH;
    const int s = ei[e], d = ei[NEDGE + e];
    const float v = edge_logit(a_src, a_dst, s, d, h);
    atomicMax(&nmax[d * NH + h], f2ord(v));
}

__global__ void edge_sum_kernel(const int* __restrict__ ei,
                                const float* __restrict__ a_src,
                                const float* __restrict__ a_dst,
                                const unsigned* __restrict__ nmax,
                                float* __restrict__ nsum)
{
    int t = blockIdx.x * blockDim.x + threadIdx.x;
    if (t >= NEDGE * NH) return;
    const int e = t / NH, h = t % NH;
    const int s = ei[e], d = ei[NEDGE + e];
    const float v = edge_logit(a_src, a_dst, s, d, h);
    const float ex = __expf(v - ord2f(nmax[d * NH + h]));
    atomicAdd(&nsum[d * NH + h], ex);
}

__global__ void edge_agg_kernel(const int* __restrict__ ei,
                                const float* __restrict__ a_src,
                                const float* __restrict__ a_dst,
                                const unsigned* __restrict__ nmax,
                                const float* __restrict__ nsum,
                                const float* __restrict__ hp,
                                float* __restrict__ hnext)
{
    int t = blockIdx.x * blockDim.x + threadIdx.x;
    if (t >= NEDGE * NH) return;
    const int e = t / NH, h = t % NH;
    const int s = ei[e], d = ei[NEDGE + e];
    const float v  = edge_logit(a_src, a_dst, s, d, h);
    const float al = __expf(v - ord2f(nmax[d * NH + h])) / nsum[d * NH + h];
    const float* sf = hp + (size_t)s * DDIM + h * NC_;
    float* df       = hnext + (size_t)d * DDIM + h * NC_;
#pragma unroll
    for (int c = 0; c < NC_; ++c) atomicAdd(&df[c], al * sf[c]);
}

__global__ void bias_elu_kernel(const float* __restrict__ acc,
                                const float* __restrict__ bias,
                                float* __restrict__ hout)
{
    int t = blockIdx.x * blockDim.x + threadIdx.x;
    if (t >= NNODE * DDIM) return;
    float v = acc[t] + bias[t % DDIM];
    hout[t] = v > 0.f ? v : (__expf(v) - 1.0f);   // ELU
}

__global__ void pool_kernel(const float* __restrict__ h,
                            const int* __restrict__ batch,
                            float* __restrict__ pooled,
                            float* __restrict__ cnt)
{
    int t = blockIdx.x * blockDim.x + threadIdx.x;
    if (t >= NNODE * DDIM) return;
    const int node = t / DDIM, d = t % DDIM;
    const int g = batch[node];
    atomicAdd(&pooled[g * DDIM + d], h[t]);
    if (d == 0) atomicAdd(&cnt[g], 1.0f);
}

__global__ void head_kernel(const float* __restrict__ pooled,
                            const float* __restrict__ cnt,
                            const float* __restrict__ out_w,
                            const float* __restrict__ out_b,
                            float* __restrict__ out)
{
    int t = blockIdx.x * blockDim.x + threadIdx.x;
    if (t >= NG * NCLS) return;
    const int g = t / NCLS, k = t % NCLS;
    const float c = fmaxf(cnt[g], 1.0f);
    float s = out_b[k];
    for (int d = 0; d < DDIM; ++d)
        s += (pooled[g * DDIM + d] / c) * out_w[d * NCLS + k];
    out[t] = s;
}

extern "C" void kernel_launch(void* const* d_in, const int* in_sizes, int n_in,
                              void* d_out, int out_size, void* d_ws, size_t ws_size,
                              hipStream_t stream)
{
    const float* x       = (const float*)d_in[0];
    const int*   ei      = (const int*)  d_in[1];   // [2,E] flat: src then dst
    const int*   batch   = (const int*)  d_in[2];
    const float* in_w    = (const float*)d_in[3];
    const float* in_b    = (const float*)d_in[4];
    const float* conv_w  = (const float*)d_in[5];   // [L,D,D]
    const float* conv_b  = (const float*)d_in[6];   // [L,D]
    const float* att_src = (const float*)d_in[7];   // [L,H,C]
    const float* att_dst = (const float*)d_in[8];
    const float* out_w   = (const float*)d_in[9];
    const float* out_b   = (const float*)d_in[10];
    float*       out     = (float*)d_out;

    // ---- workspace carve-up (floats) ----
    float* ws     = (float*)d_ws;
    float* h      = ws;                          // N*D
    float* hp     = h      + (size_t)NNODE * DDIM;
    float* hnext  = hp     + (size_t)NNODE * DDIM;
    float* a_src  = hnext  + (size_t)NNODE * DDIM;
    float* a_dst  = a_src  + (size_t)NNODE * NH;
    unsigned* nmax = (unsigned*)(a_dst + (size_t)NNODE * NH);
    float* nsum   = (float*)(nmax + (size_t)NNODE * NH);
    float* pooled = nsum   + (size_t)NNODE * NH;
    float* cnt    = pooled + (size_t)NG * DDIM;

    const int gemm_blocks = (ROW_TILES + WPB - 1) / WPB;   // 391
    const int T = 256;
    const int nEH = NEDGE * NH;
    const int nND = NNODE * DDIM;
    const int nNH = NNODE * NH;
    const unsigned neg_inf_ord = f2ord(-1e30f);            // host-side constant

    // h = x @ in_w + in_b
    gemm128_wmma_kernel<<<gemm_blocks, 256, 0, stream>>>(x, in_w, in_b, h);

    for (int l = 0; l < NLAY; ++l) {
        const float* cw = conv_w + (size_t)l * DDIM * DDIM;
        const float* cb = conv_b + (size_t)l * DDIM;
        const float* as = att_src + (size_t)l * NH * NC_;
        const float* ad = att_dst + (size_t)l * NH * NC_;

        // hp = h @ conv_w[l]  (bias applied after aggregation)
        gemm128_wmma_kernel<<<gemm_blocks, 256, 0, stream>>>(h, cw, nullptr, hp);

        att_kernel<<<(nNH + T - 1) / T, T, 0, stream>>>(hp, as, ad, a_src, a_dst);

        fill_u32_kernel<<<(nNH + T - 1) / T, T, 0, stream>>>(nmax, neg_inf_ord, nNH);
        (void)hipMemsetAsync(nsum,  0, (size_t)nNH * sizeof(float), stream);
        (void)hipMemsetAsync(hnext, 0, (size_t)nND * sizeof(float), stream);

        edge_max_kernel<<<(nEH + T - 1) / T, T, 0, stream>>>(ei, a_src, a_dst, nmax);
        edge_sum_kernel<<<(nEH + T - 1) / T, T, 0, stream>>>(ei, a_src, a_dst, nmax, nsum);
        edge_agg_kernel<<<(nEH + T - 1) / T, T, 0, stream>>>(ei, a_src, a_dst, nmax, nsum, hp, hnext);

        bias_elu_kernel<<<(nND + T - 1) / T, T, 0, stream>>>(hnext, cb, h);
    }

    (void)hipMemsetAsync(pooled, 0, (size_t)NG * DDIM * sizeof(float), stream);
    (void)hipMemsetAsync(cnt,    0, (size_t)NG * sizeof(float), stream);
    pool_kernel<<<(nND + T - 1) / T, T, 0, stream>>>(h, batch, pooled, cnt);
    head_kernel<<<(NG * NCLS + T - 1) / T, T, 0, stream>>>(pooled, cnt, out_w, out_b, out);
}
